// EMAU_10788957847959
// MI455X (gfx1250) — matrix operand order repaired and checked by reading, not compiled, not run on hardware
//
#include <hip/hip_runtime.h>
#include <hip/hip_bf16.h>
#include <stdint.h>

// ---------------------------------------------------------------------------
// EMAU block for MI455X (gfx1250), wave32 + WMMA bf16 (f32 accumulate).
// Shapes fixed by the reference: b=4, c=512, h=w=128 (n=16384), k=64.
//
// All GEMMs use the "transposed streaming" formulation:
//   C[M][N] = A[M][K] * B^T,  A row-major [M][K] (big matrix, streamed once),
//   B stored row-major [N][K] (small matrix, L2-resident, reused).
// Both operands load per-lane K-contiguous 2x16B fragments -> no LDS, no
// barriers, no staging inside any GEMM hot loop.
// The C fragment holds 8 consecutive M-rows per lane at fixed N, so a
// transposed [N][M] output is a single b128 store per tile (used to produce
// both activation layouts for free and to feed the BN epilogue directly).
// ---------------------------------------------------------------------------

typedef __attribute__((ext_vector_type(16))) __bf16 v16bf;
typedef __attribute__((ext_vector_type(8)))  float  v8f;

#define BATCH  4
#define CH     512
#define NPIX   16384   // 128*128
#define KCENT  64
#define NSPLIT 16      // split-K factor for the mu-update GEMM

__device__ __forceinline__ unsigned short f2bf(float f) {
  unsigned int u = __float_as_uint(f);
  u += 0x7FFFu + ((u >> 16) & 1u);          // round-to-nearest-even
  return (unsigned short)(u >> 16);
}
__device__ __forceinline__ float bf2f(unsigned short s) {
  return __uint_as_float(((unsigned int)s) << 16);
}

union FragAB { uint4 q[2]; v16bf v; };
union PackC  { unsigned short s[8]; uint4 q; };

// ---------------------------------------------------------------------------
// Direct-load WMMA GEMM:  C[M][N] = A[M][K] x (Bt[N][K])^T
// 256 threads = 8 waves stacked along M; each wave computes 16 x (TN*16).
// A fragment (16x32): lane m = M row; lanes 0-15 hold K 0-7 & 16-23,
//   lanes 16-31 hold K 8-15 & 24-31  -> two contiguous 16B loads.
// B fragment (32x16): lane m = N col; lanes 0-15 hold K 0-7 & 16-23, lanes 16-31
//   hold K 8-15 & 24-31 -> two contiguous 16B loads from Bt[col][k].
// blockIdx.z enumerates batch*nsplit+split; split-K partials go to
// C + bz*strideC (plain f32 stores, deterministic reduction later).
// PRIM: primary [M][N] store: 0 = none, 1 = bf16 scalar, 2 = f32 scalar.
// Ct (optional, nullptr to skip): bf16 [N][M] output, one b128 per tile.
// bias is per-COLUMN (output channel); relu applied to all stores.
// Requires: M % 128 == 0, N % (TN*16) == 0, (K/nsplit) % 32 == 0.
// ---------------------------------------------------------------------------
template<int TN, int PRIM>
__global__ __launch_bounds__(256) void gemm_direct(
    const unsigned short* __restrict__ A,  int lda, size_t strideA,
    const unsigned short* __restrict__ Bt, int ldb, size_t strideB,
    void* __restrict__ Cv, int ldc, size_t strideC,
    unsigned short* __restrict__ Ct, int ldcT, size_t strideCt,
    int M, int N, int K, int nsplit,
    const float* __restrict__ bias, int relu)
{
  constexpr int BM = 128;
  constexpr int BN = TN * 16;
  const int tid  = threadIdx.x;
  const int lane = tid & 31;
  const int w    = tid >> 5;
  const int bz    = blockIdx.z;
  const int batch = bz / nsplit;
  const int split = bz % nsplit;
  const int kchunk = K / nsplit;
  const int kBeg   = split * kchunk;
  const int steps  = kchunk >> 5;

  const unsigned short* Ab  = A  + (size_t)batch * strideA;
  const unsigned short* Btb = Bt + (size_t)batch * strideB;

  const int row0    = blockIdx.x * BM + w * 16;
  const int colBase = blockIdx.y * BN;
  const int m  = lane & 15;
  const int kh = lane >> 4;

  const unsigned short* ap  = Ab  + (size_t)(row0 + m) * lda + kBeg + 8 * kh;
  const unsigned short* bp0 = Btb + (size_t)(colBase + m) * ldb + kBeg + 8 * kh;
  const size_t btile = (size_t)16 * ldb;

  v8f acc[TN];
  v8f zero = {0.f,0.f,0.f,0.f,0.f,0.f,0.f,0.f};
  #pragma unroll
  for (int t = 0; t < TN; t++) acc[t] = zero;

  for (int i = 0; i < steps; i++, ap += 32, bp0 += 32) {
    FragAB fa;
    fa.q[0] = *(const uint4*)(ap);
    fa.q[1] = *(const uint4*)(ap + 16);
    #pragma unroll
    for (int t = 0; t < TN; t++) {
      FragAB fb;
      const unsigned short* bp = bp0 + (size_t)t * btile;
      fb.q[0] = *(const uint4*)(bp);
      fb.q[1] = *(const uint4*)(bp + 16);
      acc[t] = __builtin_amdgcn_wmma_f32_16x16x32_bf16(
          false, fa.v, false, fb.v, (short)0, acc[t], false, false);
    }
  }

  // --- C fragment: VGPR r -> M = rowS + r (rowS = row0 + 8*kh), N = col ---
  const int rowS = row0 + kh * 8;
  if (PRIM == 1) {
    unsigned short* C = (unsigned short*)Cv + (size_t)bz * strideC;
    #pragma unroll
    for (int t = 0; t < TN; t++) {
      const int col = colBase + t * 16 + m;
      float bc = bias ? bias[col] : 0.f;
      #pragma unroll
      for (int r = 0; r < 8; r++) {
        float v = acc[t][r] + bc;
        if (relu) v = fmaxf(v, 0.f);
        C[(size_t)(rowS + r) * ldc + col] = f2bf(v);
      }
    }
  } else if (PRIM == 2) {
    float* C = (float*)Cv + (size_t)bz * strideC;
    #pragma unroll
    for (int t = 0; t < TN; t++) {
      const int col = colBase + t * 16 + m;
      float bc = bias ? bias[col] : 0.f;
      #pragma unroll
      for (int r = 0; r < 8; r++) {
        float v = acc[t][r] + bc;
        if (relu) v = fmaxf(v, 0.f);
        C[(size_t)(rowS + r) * ldc + col] = v;
      }
    }
  }
  if (Ct) {  // packed transposed store: one b128 per tile
    unsigned short* Cb = Ct + (size_t)batch * strideCt;
    #pragma unroll
    for (int t = 0; t < TN; t++) {
      const int col = colBase + t * 16 + m;
      float bc = bias ? bias[col] : 0.f;
      PackC pc;
      #pragma unroll
      for (int r = 0; r < 8; r++) {
        float v = acc[t][r] + bc;
        if (relu) v = fmaxf(v, 0.f);
        pc.s[r] = f2bf(v);
      }
      *(uint4*)(Cb + (size_t)col * ldcT + rowS) = pc.q;
    }
  }
}

// ---------------------------------------------------------------------------
// Transpose + convert: x [b][CH][NPIX] f32 -> xt [b][NPIX][CH] bf16
// 32x32 tiles through padded LDS; both sides coalesced.
// ---------------------------------------------------------------------------
__global__ __launch_bounds__(256) void tcvt_kernel(
    const float* __restrict__ src, unsigned short* __restrict__ dst)
{
  __shared__ float tile[32][33];
  const int b  = blockIdx.z;
  const int bn = blockIdx.x * 32;   // NPIX tile
  const int bc = blockIdx.y * 32;   // CH tile
  const int tc = threadIdx.x & 31;
  const int tr = threadIdx.x >> 5;  // 8 rows per pass
  const float* s = src + ((size_t)b * CH + bc) * NPIX + bn;
  #pragma unroll
  for (int p = 0; p < 4; p++)
    tile[tr + p * 8][tc] = s[(size_t)(tr + p * 8) * NPIX + tc];
  __syncthreads();
  unsigned short* d = dst + ((size_t)b * NPIX + bn) * CH + bc;
  #pragma unroll
  for (int p = 0; p < 4; p++)
    d[(size_t)(tr + p * 8) * CH + tc] = f2bf(tile[tc][tr + p * 8]);
}

// fp32 -> bf16 conversion, 4 elements/thread (weights, kept [o][c])
__global__ __launch_bounds__(256) void cvt4_kernel(
    const float4* __restrict__ src, ushort4* __restrict__ dst, int count4)
{
  int g = blockIdx.x * 256 + threadIdx.x;
  if (g < count4) {
    float4 f = src[g];
    ushort4 o;
    o.x = f2bf(f.x); o.y = f2bf(f.y); o.z = f2bf(f.z); o.w = f2bf(f.w);
    dst[g] = o;
  }
}

// mu0 [1][c][k] fp32 -> mub [b][c][k] bf16 (broadcast) and mutb [b][k][c] bf16
__global__ __launch_bounds__(256) void mu_init_kernel(
    const float* __restrict__ mu0,
    unsigned short* __restrict__ mub, unsigned short* __restrict__ mutb)
{
  int g = blockIdx.x * 256 + threadIdx.x;          // [b][c][k]
  int kk = g & (KCENT - 1);
  int c  = (g >> 6) & (CH - 1);
  int b  = g >> 15;
  unsigned short v = f2bf(mu0[(c << 6) + kk]);
  mub[g] = v;
  mutb[((size_t)b * KCENT + kk) * CH + c] = v;
}

// ---------------------------------------------------------------------------
// Softmax over k=64 per pixel. zlog is [b][n][64] fp32: per-thread rows are
// CONTIGUOUS (b128 loads). Writes z [b][n][64] bf16 (contiguous b64 stores)
// and zt [b][64][n] bf16 (strided, B operand of mu-update GEMM).
// ---------------------------------------------------------------------------
__global__ __launch_bounds__(256) void softmax_kernel(
    const float* __restrict__ zlog,
    unsigned short* __restrict__ z, unsigned short* __restrict__ zt)
{
  int g = blockIdx.x * 256 + threadIdx.x;          // over BATCH*NPIX
  int b = g >> 14;
  int j = g & (NPIX - 1);
  const float4* p = (const float4*)(zlog + (size_t)g * KCENT);
  float v[KCENT];
  float mx = -1e30f;
  #pragma unroll
  for (int q = 0; q < KCENT / 4; q++) {
    float4 f = p[q];
    v[4*q]   = f.x; v[4*q+1] = f.y; v[4*q+2] = f.z; v[4*q+3] = f.w;
  }
  #pragma unroll
  for (int i = 0; i < KCENT; i++) mx = fmaxf(mx, v[i]);
  float s = 0.f;
  #pragma unroll
  for (int i = 0; i < KCENT; i++) { v[i] = __expf(v[i] - mx); s += v[i]; }
  float inv = 1.f / s;
  ushort4* zr = (ushort4*)(z + (size_t)g * KCENT);
  #pragma unroll
  for (int q = 0; q < KCENT / 4; q++) {
    ushort4 o;
    o.x = f2bf(v[4*q]   * inv);
    o.y = f2bf(v[4*q+1] * inv);
    o.z = f2bf(v[4*q+2] * inv);
    o.w = f2bf(v[4*q+3] * inv);
    zr[q] = o;
  }
  #pragma unroll
  for (int i = 0; i < KCENT; i++)
    zt[((size_t)b * KCENT + i) * NPIX + j] = f2bf(v[i] * inv);
}

// ---------------------------------------------------------------------------
// Reduce split-K partials [b][NSPLIT][c][k], l2-normalize each column over c
// (the z colsum scaling cancels inside the norm, so it is skipped), write
// mub [b][c][k] bf16 + mutb [b][k][c] bf16 (+ final fp32 mu on last stage).
// ---------------------------------------------------------------------------
__global__ __launch_bounds__(256) void mu_norm_kernel(
    const float* __restrict__ muacc,
    unsigned short* __restrict__ mub, unsigned short* __restrict__ mutb,
    float* __restrict__ muout)
{
  int b  = blockIdx.x >> 6;
  int kk = blockIdx.x & (KCENT - 1);
  int c0 = threadIdx.x;
  int c1 = threadIdx.x + 256;
  float a0 = 0.f, a1 = 0.f;
  for (int s = 0; s < NSPLIT; s++) {
    size_t base = ((size_t)b * NSPLIT + s) * CH * KCENT + kk;
    a0 += muacc[base + (size_t)c0 * KCENT];
    a1 += muacc[base + (size_t)c1 * KCENT];
  }
  __shared__ float red[256];
  red[threadIdx.x] = a0 * a0 + a1 * a1;
  __syncthreads();
  for (int o = 128; o > 0; o >>= 1) {
    if (threadIdx.x < o) red[threadIdx.x] += red[threadIdx.x + o];
    __syncthreads();
  }
  float scale = 1.f / (1e-6f + sqrtf(red[0]));
  float n0 = a0 * scale, n1 = a1 * scale;
  mub[((size_t)b * CH + c0) * KCENT + kk] = f2bf(n0);
  mub[((size_t)b * CH + c1) * KCENT + kk] = f2bf(n1);
  mutb[((size_t)b * KCENT + kk) * CH + c0] = f2bf(n0);
  mutb[((size_t)b * KCENT + kk) * CH + c1] = f2bf(n1);
  if (muout) {
    muout[((size_t)b * CH + c0) * KCENT + kk] = n0;
    muout[((size_t)b * CH + c1) * KCENT + kk] = n1;
  }
}

// ---------------------------------------------------------------------------
// BatchNorm batch statistics over y [b][c][n]: one block per channel,
// fully coalesced b64 reads, deterministic block-tree reduction.
// ---------------------------------------------------------------------------
__global__ __launch_bounds__(256) void bn_stats_kernel(
    const unsigned short* __restrict__ y, float* __restrict__ stats)
{
  int c = blockIdx.x;
  float s = 0.f, sq = 0.f;
  for (int b = 0; b < BATCH; b++) {
    const ushort4* p = (const ushort4*)(y + ((size_t)b * CH + c) * NPIX);
    for (int i = threadIdx.x; i < NPIX / 4; i += 256) {
      ushort4 u = p[i];
      float v0 = bf2f(u.x), v1 = bf2f(u.y), v2 = bf2f(u.z), v3 = bf2f(u.w);
      s  += (v0 + v1) + (v2 + v3);
      sq += (v0 * v0 + v1 * v1) + (v2 * v2 + v3 * v3);
    }
  }
  __shared__ float rs[256];
  __shared__ float rq[256];
  rs[threadIdx.x] = s;
  rq[threadIdx.x] = sq;
  __syncthreads();
  for (int o = 128; o > 0; o >>= 1) {
    if (threadIdx.x < o) { rs[threadIdx.x] += rs[threadIdx.x + o];
                           rq[threadIdx.x] += rq[threadIdx.x + o]; }
    __syncthreads();
  }
  if (threadIdx.x == 0) {
    const float inv = 1.f / (float)(BATCH * NPIX);
    float mean = rs[0] * inv;
    float var  = rq[0] * inv - mean * mean;
    stats[2 * c]     = mean;
    stats[2 * c + 1] = var;
  }
}

// out = relu( (y - mean)*rsqrt(var+eps)*bn_w + bn_b + x ), fp32, 4 elem/thread
__global__ __launch_bounds__(256) void final_kernel(
    const ushort4* __restrict__ y, const float4* __restrict__ x,
    const float* __restrict__ stats, const float* __restrict__ bnw,
    const float* __restrict__ bnb, float4* __restrict__ out)
{
  size_t g = (size_t)blockIdx.x * 256 + threadIdx.x;   // over b*c*n/4
  int c = (int)((g >> 12) & (CH - 1));                 // n/4 = 4096 = 2^12
  float mean = stats[2 * c], var = stats[2 * c + 1];
  float sc = rsqrtf(var + 1e-5f) * bnw[c];
  float of = bnb[c] - mean * sc;
  ushort4 u = y[g];
  float4 xv = x[g];
  float4 o;
  o.x = fmaxf(bf2f(u.x) * sc + of + xv.x, 0.f);
  o.y = fmaxf(bf2f(u.y) * sc + of + xv.y, 0.f);
  o.z = fmaxf(bf2f(u.z) * sc + of + xv.z, 0.f);
  o.w = fmaxf(bf2f(u.w) * sc + of + xv.w, 0.f);
  out[g] = o;
}

// ---------------------------------------------------------------------------
// Host launcher
// ---------------------------------------------------------------------------
extern "C" void kernel_launch(void* const* d_in, const int* in_sizes, int n_in,
                              void* d_out, int out_size, void* d_ws, size_t ws_size,
                              hipStream_t stream) {
  const float* x   = (const float*)d_in[0];
  const float* w1  = (const float*)d_in[1];
  const float* b1  = (const float*)d_in[2];
  const float* w2  = (const float*)d_in[3];
  const float* bnw = (const float*)d_in[4];
  const float* bnb = (const float*)d_in[5];
  const float* mu0 = (const float*)d_in[6];
  (void)in_sizes; (void)n_in; (void)out_size; (void)ws_size;

  char* ws = (char*)d_ws;
  unsigned short* xt    = (unsigned short*)(ws + 0);           //  64 MiB  x^T  bf16 [b][n][c]
  unsigned short* x1t   = (unsigned short*)(ws + 67108864);    //  64 MiB  x1^T bf16 [b][n][c]
  unsigned short* x1    = (unsigned short*)(ws + 134217728);   //  64 MiB  x1   bf16 [b][c][n]
  unsigned short* w1b   = (unsigned short*)(ws + 201326592);   // 512 KiB  [o][c]
  unsigned short* w2b   = (unsigned short*)(ws + 201850880);   // 512 KiB  [o][c]
  unsigned short* mub   = (unsigned short*)(ws + 202375168);   // 256 KiB  [b][c][k]
  unsigned short* mutb  = (unsigned short*)(ws + 202637312);   // 256 KiB  [b][k][c]
  float*          zlog  = (float*)         (ws + 202899456);   //  16 MiB  [b][n][k]
  unsigned short* z     = (unsigned short*)(ws + 219676672);   //   8 MiB  [b][n][k]
  unsigned short* zt    = (unsigned short*)(ws + 228065280);   //   8 MiB  [b][k][n]
  float*          muacc = (float*)         (ws + 236453888);   //   8 MiB  [b][NSPLIT][c][k]
  float*          stats = (float*)         (ws + 244842496);   //   4 KiB
  unsigned short* xrect = xt;    // x^T dead after conv1
  unsigned short* y     = x1t;   // x1^T dead after last EM stage; y bf16 [b][c][n]

  float* outp  = (float*)d_out;
  float* muout = outp + (size_t)BATCH * CH * NPIX;             // mu is 2nd output

  const size_t sAct = (size_t)CH * NPIX;       // per-batch activation
  const size_t sMu  = (size_t)CH * KCENT;      // per-batch [c][k] / [k][c]
  const size_t sZ   = (size_t)NPIX * KCENT;    // per-batch [n][k] / [k][n]

  // transpose-convert x, convert weights, init mu
  tcvt_kernel<<<dim3(512,16,BATCH), 256, 0, stream>>>(x, xt);
  cvt4_kernel<<<256, 256, 0, stream>>>((const float4*)w1, (ushort4*)w1b, CH*CH/4);
  cvt4_kernel<<<256, 256, 0, stream>>>((const float4*)w2, (ushort4*)w2b, CH*CH/4);
  mu_init_kernel<<<512, 256, 0, stream>>>(mu0, mub, mutb);

  // conv1 (M=NPIX, N=512, K=512): x1^T[n][o] scalar + x1[o][n] packed b128
  gemm_direct<8,1><<<dim3(128,4,BATCH), 256, 0, stream>>>(
      xt, CH, sAct, w1b, CH, 0,
      x1t, CH, sAct, x1, NPIX, sAct,
      NPIX, CH, CH, 1, b1, 0);

  for (int s = 0; s < 3; s++) {
    // logits[n][k] = x1^T * mu   (M=NPIX, N=64, K=512), f32 out
    gemm_direct<4,2><<<dim3(128,1,BATCH), 256, 0, stream>>>(
        x1t, CH, sAct, mutb, CH, sMu,
        zlog, KCENT, sZ, nullptr, 0, 0,
        NPIX, KCENT, CH, 1, nullptr, 0);
    // softmax over k: z [n][k] and z^T [k][n]
    softmax_kernel<<<256, 256, 0, stream>>>(zlog, z, zt);
    // mu_acc[c][k] = x1 * z      (M=512, N=64, K=NPIX), 16-way split-K
    gemm_direct<4,2><<<dim3(4,1,BATCH*NSPLIT), 256, 0, stream>>>(
        x1, NPIX, sAct, zt, NPIX, sZ,
        muacc, KCENT, sMu, nullptr, 0, 0,
        CH, KCENT, NPIX, NSPLIT, nullptr, 0);
    // reduce + l2norm -> mub/mutb (+ fp32 mu output on last stage)
    mu_norm_kernel<<<256, 256, 0, stream>>>(
        muacc, mub, mutb, (s == 2) ? muout : nullptr);
  }

  // xrec^T[n][c] = relu(z * mu^T)   (M=NPIX, N=512, K=64)
  gemm_direct<8,1><<<dim3(128,4,BATCH), 256, 0, stream>>>(
      z, KCENT, sZ, mub, KCENT, sMu,
      xrect, CH, sAct, nullptr, 0, 0,
      NPIX, CH, KCENT, 1, nullptr, 1);

  // conv2 (M=NPIX, N=512, K=512): store ONLY y[c][n] via packed b128
  gemm_direct<8,0><<<dim3(128,4,BATCH), 256, 0, stream>>>(
      xrect, CH, sAct, w2b, CH, 0,
      nullptr, 0, 0, y, NPIX, sAct,
      NPIX, CH, CH, 1, nullptr, 0);

  // BN stats + fused BN/residual/relu epilogue (all coalesced on y[c][n])
  bn_stats_kernel<<<512, 256, 0, stream>>>(y, stats);
  final_kernel<<<32768, 256, 0, stream>>>((const ushort4*)y, (const float4*)x,
                                          stats, bnw, bnb, (float4*)outp);
}
